// Generalised_GMoE_76227079569517
// MI455X (gfx1250) — compile-verified, matrix operand order checked by reading, not run on hardware
//
#include <hip/hip_runtime.h>

// ---------------------------------------------------------------------------
// CDNA5 (gfx1250) MoE U-Net: conv-as-implicit-GEMM via v_wmma_f32_16x16x32_f16
// Async (ASYNCcnt) global->LDS weight staging, 2 N-tiles per wave,
// fully unrolled K loop.
// ---------------------------------------------------------------------------

typedef __attribute__((ext_vector_type(16))) _Float16 v16h;
typedef __attribute__((ext_vector_type(8)))  _Float16 v8h;
typedef __attribute__((ext_vector_type(8)))  float    v8f;

__device__ __forceinline__ int kmap(int j, int h) {
    // 16-bit A/B fragment: element j (0..15) of lane-half h (0/1) -> K index
    // (ISA 7.12.2: VGPR0..3 hold K=0..7 / K=8..15 per half, VGPR4..7 hold +16)
    return j + h * 8 + ((j >= 8) ? 8 : 0);
}

// ---------------- weight packing: (E,Cout,Cin,3,3) f32 -> f16 [E][Mpad][Kpad]
__global__ void pack_w_kernel(const float* __restrict__ cw, _Float16* __restrict__ Ap,
                              int Cout, int Cin, int Mpad, int Kpad) {
    int e = blockIdx.x;
    int K = Cin * 9;
    for (int idx = threadIdx.x; idx < Mpad * Kpad; idx += blockDim.x) {
        int m = idx / Kpad, k = idx - m * Kpad;
        float v = 0.f;
        if (m < Cout && k < K) v = cw[((size_t)e * Cout + m) * K + k];
        Ap[((size_t)e * Mpad + m) * Kpad + k] = (_Float16)v;
    }
}

// ---------------- gate: graw[b,e] = (xf@gw+gb) + eps*softplus(xf@nw+nb)
__global__ void gate_raw_kernel(const float* __restrict__ xf,
                                const float* __restrict__ gw, const float* __restrict__ gb,
                                const float* __restrict__ nw, const float* __restrict__ nb,
                                const float* __restrict__ eps,
                                float* __restrict__ graw, int D) {
    __shared__ float red[8][256];
    int b = blockIdx.x, tid = threadIdx.x;
    float acc[8];
    #pragma unroll
    for (int q = 0; q < 8; ++q) acc[q] = 0.f;
    const float* row = xf + (size_t)b * D;
    for (int d = tid; d < D; d += 256) {
        float xv = row[d];
        #pragma unroll
        for (int e = 0; e < 4; ++e) {
            acc[e]     += xv * gw[(size_t)d * 4 + e];
            acc[4 + e] += xv * nw[(size_t)d * 4 + e];
        }
    }
    #pragma unroll
    for (int q = 0; q < 8; ++q) red[q][tid] = acc[q];
    __syncthreads();
    for (int s = 128; s > 0; s >>= 1) {
        if (tid < s) {
            #pragma unroll
            for (int q = 0; q < 8; ++q) red[q][tid] += red[q][tid + s];
        }
        __syncthreads();
    }
    if (tid < 4) {
        int e = tid;
        float g  = red[e][0] + gb[e];
        float nv = red[4 + e][0] + nb[e];
        float sp = (nv > 0.f) ? (nv + log1pf(expf(-nv))) : log1pf(expf(nv));
        graw[(size_t)b * 4 + e] = g + eps[e] * sp;
    }
}

// ---------------- top-k over batch-summed gates -> mask[4] in {0,1}
__global__ void gate_topk_kernel(const float* __restrict__ graw,
                                 const int* __restrict__ kptr,
                                 float* __restrict__ mask, int B) {
    __shared__ float red[4][256];
    int tid = threadIdx.x;
    float acc[4] = {0.f, 0.f, 0.f, 0.f};
    for (int b = tid; b < B; b += 256) {
        #pragma unroll
        for (int e = 0; e < 4; ++e) acc[e] += graw[(size_t)b * 4 + e];
    }
    #pragma unroll
    for (int e = 0; e < 4; ++e) red[e][tid] = acc[e];
    __syncthreads();
    for (int s = 128; s > 0; s >>= 1) {
        if (tid < s) {
            #pragma unroll
            for (int e = 0; e < 4; ++e) red[e][tid] += red[e][tid + s];
        }
        __syncthreads();
    }
    if (tid == 0) {
        int kk = *kptr;
        if (kk > 4) kk = 4;
        if (kk < 0) kk = 0;
        float cs[4];
        bool  sel[4] = {false, false, false, false};
        #pragma unroll
        for (int e = 0; e < 4; ++e) cs[e] = red[e][0];
        for (int t = 0; t < kk; ++t) {
            int best = -1; float bv = -3.4e38f;
            for (int e = 0; e < 4; ++e)
                if (!sel[e] && cs[e] > bv) { bv = cs[e]; best = e; }
            if (best >= 0) sel[best] = true;
        }
        for (int e = 0; e < 4; ++e) mask[e] = sel[e] ? 1.f : 0.f;
    }
}

// ---------------- masked (zero BEFORE softmax) row softmax
__global__ void gate_softmax_kernel(const float* __restrict__ graw,
                                    const float* __restrict__ mask,
                                    float* __restrict__ gate, int B) {
    int b = blockIdx.x * blockDim.x + threadIdx.x;
    if (b >= B) return;
    float g[4], mx = -3.4e38f;
    #pragma unroll
    for (int e = 0; e < 4; ++e) { g[e] = graw[(size_t)b * 4 + e] * mask[e]; mx = fmaxf(mx, g[e]); }
    float s = 0.f;
    #pragma unroll
    for (int e = 0; e < 4; ++e) { g[e] = expf(g[e] - mx); s += g[e]; }
    float inv = 1.f / s;
    #pragma unroll
    for (int e = 0; e < 4; ++e) gate[(size_t)b * 4 + e] = g[e] * inv;
}

// ---------------- im2col gather of one B-fragment column (16 K-values)
template <int CIN>
__device__ __forceinline__ v16h gather_bfrag(const float* __restrict__ xb,
                                             int H, int W, int oh, int ow,
                                             bool colv, int kb, int h16) {
    constexpr int K = CIN * 9;
    v16h bf;
    #pragma unroll
    for (int j = 0; j < 16; ++j) {
        int k = kb + kmap(j, h16);
        float bv = 0.f;
        if (k < K) {
            int ci = k / 9, r = k - ci * 9;
            int kh = r / 3, kw = r - kh * 3;
            int ih = oh + kh - 1, iw = ow + kw - 1;
            if (colv && (unsigned)ih < (unsigned)H && (unsigned)iw < (unsigned)W)
                bv = xb[((size_t)ci * H + ih) * W + iw];
        }
        bf[j] = (_Float16)bv;
    }
    return bf;
}

// ---------------- fused MoE conv: D = sum_e gate[b,e]*relu(W_e (x) im2col + b_e)
// Block = 8 waves; expert weights async-staged into LDS; each wave owns a
// 16(M) x 32(N) slab = two 16x16 WMMA tiles sharing every A fragment.
template <int CIN>
__global__ __launch_bounds__(256)
void moe_conv_wmma_kernel(const float* __restrict__ x, const _Float16* __restrict__ Ap,
                          const float* __restrict__ bias, const float* __restrict__ gate,
                          float* __restrict__ out,
                          int B, int H, int W, int Cout, int Mtiles) {
    constexpr int E    = 4;
    constexpr int K    = CIN * 9;
    constexpr int KPAD = ((K + 31) / 32) * 32;

    extern __shared__ _Float16 smem[];     // [E][Mpad][KPAD] f16 weights
    int tid  = threadIdx.x;
    int lane = tid & 31;
    int wv   = tid >> 5;
    int Mpad = Mtiles * 16;

    // ---- stage all experts' packed weights into LDS.
    // tot32 = 4*Mpad*KPAD/2 is a multiple of 256 for every layer, so the loop
    // has no ragged tail: EXEC is all-1s at each async issue.
    {
        int tot32 = (E * Mpad * KPAD) >> 1;             // f16 pairs (dwords)
        const unsigned* gsrc = (const unsigned*)Ap;
        unsigned*       gdst = (unsigned*)smem;
#if defined(__gfx1250__)
        for (int i = tid; i < tot32; i += 256) {
            // GV-mode async DMA: LDS[vdst] = MEM[vaddr]; tracked by ASYNCcnt.
            // LDS aperture maps flat addr[31:0] -> LDS byte address.
            unsigned       ldsb = (unsigned)(unsigned long long)(gdst + i);
            const unsigned* ga  = gsrc + i;
            asm volatile("global_load_async_to_lds_b32 %0, %1, off"
                         :: "v"(ldsb), "v"(ga) : "memory");
        }
        asm volatile("s_wait_asynccnt 0x0" ::: "memory");
#else
        for (int i = tid; i < tot32; i += 256) gdst[i] = gsrc[i];
#endif
    }
    __syncthreads();
#if defined(__gfx1250__)
    __builtin_amdgcn_s_wait_tensorcnt(0);  // gfx1250 TENSORcnt wait (no-op here)
#endif

    long long total = (long long)B * H * W;
    long long pairs = (total + 31) >> 5;   // 32-column slabs
    long long tiles = pairs * Mtiles;
    long long tile  = (long long)blockIdx.x * 8 + wv;
    if (tile >= tiles) return;             // wave-uniform: EXEC stays all-1s

    int       mt = (int)(tile % Mtiles);
    long long pt = tile / Mtiles;

    int h16 = lane >> 4;                   // lane half (K striping)
    int l16 = lane & 15;                   // A row / B,C,D column

    // decode the two column sets this wave owns
    long long n0 = pt * 32 + l16;
    long long n1 = n0 + 16;
    bool colv0 = (n0 < total);
    bool colv1 = (n1 < total);
    long long c0 = colv0 ? n0 : (total - 1);
    long long c1 = colv1 ? n1 : (total - 1);
    int ow0 = (int)(c0 % W); long long t0 = c0 / W;
    int oh0 = (int)(t0 % H); int bb0 = (int)(t0 / H);
    int ow1 = (int)(c1 % W); long long t1 = c1 / W;
    int oh1 = (int)(t1 % H); int bb1 = (int)(t1 / H);

    const float* xb0 = x + (size_t)bb0 * CIN * H * W;
    const float* xb1 = x + (size_t)bb1 * CIN * H * W;
    __builtin_prefetch(xb0, 0, 1);         // global_prefetch_b8
    __builtin_prefetch(xb1, 0, 1);

    int row = mt * 16 + l16;               // A-matrix row (Cout index, padded)

    v8f oacc0 = {0.f, 0.f, 0.f, 0.f, 0.f, 0.f, 0.f, 0.f};
    v8f oacc1 = {0.f, 0.f, 0.f, 0.f, 0.f, 0.f, 0.f, 0.f};

    #pragma unroll 1
    for (int e = 0; e < E; ++e) {
        v8f acc0 = {0.f, 0.f, 0.f, 0.f, 0.f, 0.f, 0.f, 0.f};
        v8f acc1 = {0.f, 0.f, 0.f, 0.f, 0.f, 0.f, 0.f, 0.f};
        const _Float16* arow = smem + ((size_t)e * Mpad + row) * KPAD;
        #pragma unroll
        for (int kb = 0; kb < KPAD; kb += 32) {
            // A fragment: two contiguous 8xf16 runs per lane-half (ds_load_b128)
            const _Float16* ap = arow + kb + 8 * h16;
            v8h a0 = *(const v8h*)(ap);
            v8h a1 = *(const v8h*)(ap + 16);
            v16h afrag = __builtin_shufflevector(a0, a1,
                0, 1, 2, 3, 4, 5, 6, 7, 8, 9, 10, 11, 12, 13, 14, 15);

            v16h bf0 = gather_bfrag<CIN>(xb0, H, W, oh0, ow0, colv0, kb, h16);
            v16h bf1 = gather_bfrag<CIN>(xb1, H, W, oh1, ow1, colv1, kb, h16);

            acc0 = __builtin_amdgcn_wmma_f32_16x16x32_f16(
                false, afrag, false, bf0, (short)0, acc0, false, false);
            acc1 = __builtin_amdgcn_wmma_f32_16x16x32_f16(
                false, afrag, false, bf1, (short)0, acc1, false, false);
        }
        float g0 = gate[(size_t)bb0 * 4 + e];
        float g1 = gate[(size_t)bb1 * 4 + e];
        #pragma unroll
        for (int r = 0; r < 8; ++r) {
            int co = mt * 16 + r + h16 * 8;    // C/D layout: M = r + 8*(lane/16)
            if (co < Cout) {
                float bse = bias[e * Cout + co];
                oacc0[r] += g0 * fmaxf(acc0[r] + bse, 0.f);
                oacc1[r] += g1 * fmaxf(acc1[r] + bse, 0.f);
            }
        }
    }
    #pragma unroll
    for (int r = 0; r < 8; ++r) {
        int co = mt * 16 + r + h16 * 8;
        if (co < Cout) {
            if (colv0) out[(((size_t)bb0 * Cout + co) * H + oh0) * W + ow0] = oacc0[r];
            if (colv1) out[(((size_t)bb1 * Cout + co) * H + oh1) * W + ow1] = oacc1[r];
        }
    }
}

// ---------------- 2x2 max pool, stride 2
__global__ void maxpool2_kernel(const float* __restrict__ in, float* __restrict__ out,
                                int B, int C, int H, int W) {
    int Ho = H / 2, Wo = W / 2;
    size_t tot = (size_t)B * C * Ho * Wo;
    size_t i = (size_t)blockIdx.x * blockDim.x + threadIdx.x;
    if (i >= tot) return;
    int wo = (int)(i % Wo); size_t t = i / Wo;
    int ho = (int)(t % Ho); t /= Ho;
    int c  = (int)(t % C);  int b = (int)(t / C);
    const float* p = in + (((size_t)b * C + c) * H + 2 * ho) * W + 2 * wo;
    out[i] = fmaxf(fmaxf(p[0], p[1]), fmaxf(p[W], p[W + 1]));
}

// ---------------- bilinear x2 upsample (align_corners=True) into concat slot
__global__ void upsample2_kernel(const float* __restrict__ in, float* __restrict__ out,
                                 int B, int C, int H, int W, int Ctot, int coff) {
    int Ho = 2 * H, Wo = 2 * W;
    size_t tot = (size_t)B * C * Ho * Wo;
    size_t i = (size_t)blockIdx.x * blockDim.x + threadIdx.x;
    if (i >= tot) return;
    int ox = (int)(i % Wo); size_t t = i / Wo;
    int oy = (int)(t % Ho); t /= Ho;
    int c  = (int)(t % C);  int b = (int)(t / C);
    float fy = oy * (float)(H - 1) / (float)(Ho - 1);
    float fx = ox * (float)(W - 1) / (float)(Wo - 1);
    int y0 = (int)floorf(fy); int y1 = (y0 + 1 < H) ? y0 + 1 : H - 1; float wy = fy - y0;
    int x0 = (int)floorf(fx); int x1 = (x0 + 1 < W) ? x0 + 1 : W - 1; float wx = fx - x0;
    const float* p = in + ((size_t)b * C + c) * H * W;
    float v = (1.f - wy) * ((1.f - wx) * p[y0 * W + x0] + wx * p[y0 * W + x1]) +
              wy         * ((1.f - wx) * p[y1 * W + x0] + wx * p[y1 * W + x1]);
    out[(((size_t)b * Ctot + coff + c) * Ho + oy) * Wo + ox] = v;
}

// ---------------- copy channels into concat slot
__global__ void copy_ch_kernel(const float* __restrict__ in, float* __restrict__ out,
                               int B, int C, int H, int W, int Ctot, int coff) {
    size_t tot = (size_t)B * C * H * W;
    size_t i = (size_t)blockIdx.x * blockDim.x + threadIdx.x;
    if (i >= tot) return;
    int w = (int)(i % W); size_t t = i / W;
    int h = (int)(t % H); t /= H;
    int c = (int)(t % C); int b = (int)(t / C);
    out[(((size_t)b * Ctot + coff + c) * H + h) * W + w] = in[i];
}

// ---------------- final 1x1 conv: out[b,0,hw] = sum_c h[b,c,hw]*wl[c] + bl
__global__ void final_proj_kernel(const float* __restrict__ hbuf,
                                  const float* __restrict__ wl, const float* __restrict__ bl,
                                  float* __restrict__ out, int B, int C, int HW) {
    size_t tot = (size_t)B * HW;
    size_t i = (size_t)blockIdx.x * blockDim.x + threadIdx.x;
    if (i >= tot) return;
    int p = (int)(i % HW);
    int b = (int)(i / HW);
    float s = bl[0];
    for (int c = 0; c < C; ++c)
        s += hbuf[((size_t)b * C + c) * HW + p] * wl[c];
    out[i] = s;
}

// ===========================================================================

static inline size_t align_up(size_t v) { return (v + 255) & ~(size_t)255; }

struct MoeScratch {
    _Float16* Ap;
    float* graw;
    float* mask;
    float* gate;
};

template <int CIN>
static void launch_conv(const float* xin, const _Float16* Ap, const float* cb,
                        const float* gate, float* out,
                        int B, int H, int W, int Cout, hipStream_t stream) {
    constexpr int K    = CIN * 9;
    constexpr int KPAD = ((K + 31) / 32) * 32;
    int Mtiles = (Cout + 15) / 16;
    long long total = (long long)B * H * W;
    long long pairs = (total + 31) >> 5;
    long long tiles = pairs * Mtiles;
    int blocks = (int)((tiles + 7) / 8);
    size_t smem = (size_t)4 * (Mtiles * 16) * KPAD * sizeof(_Float16);
    moe_conv_wmma_kernel<CIN><<<blocks, 256, smem, stream>>>(
        xin, Ap, cb, gate, out, B, H, W, Cout, Mtiles);
}

static void run_moe_layer(const float* xin, int B, int Cin, int H, int W, int Cout,
                          void* const* d_in, int base, const int* kptr,
                          float* out, MoeScratch s, hipStream_t stream) {
    const float* cw  = (const float*)d_in[base + 0];
    const float* cb  = (const float*)d_in[base + 1];
    const float* gw  = (const float*)d_in[base + 2];
    const float* gb  = (const float*)d_in[base + 3];
    const float* nw  = (const float*)d_in[base + 4];
    const float* nb  = (const float*)d_in[base + 5];
    const float* eps = (const float*)d_in[base + 6];

    int K      = Cin * 9;
    int Kpad   = ((K + 31) / 32) * 32;
    int Mtiles = (Cout + 15) / 16;
    int D      = Cin * H * W;

    pack_w_kernel<<<4, 256, 0, stream>>>(cw, s.Ap, Cout, Cin, Mtiles * 16, Kpad);
    gate_raw_kernel<<<B, 256, 0, stream>>>(xin, gw, gb, nw, nb, eps, s.graw, D);
    gate_topk_kernel<<<1, 256, 0, stream>>>(s.graw, kptr, s.mask, B);
    gate_softmax_kernel<<<(B + 255) / 256, 256, 0, stream>>>(s.graw, s.mask, s.gate, B);

    switch (Cin) {
        case 1:  launch_conv<1 >(xin, s.Ap, cb, s.gate, out, B, H, W, Cout, stream); break;
        case 10: launch_conv<10>(xin, s.Ap, cb, s.gate, out, B, H, W, Cout, stream); break;
        case 20: launch_conv<20>(xin, s.Ap, cb, s.gate, out, B, H, W, Cout, stream); break;
        case 30: launch_conv<30>(xin, s.Ap, cb, s.gate, out, B, H, W, Cout, stream); break;
        default: break; // not reachable for this model
    }
}

extern "C" void kernel_launch(void* const* d_in, const int* in_sizes, int n_in,
                              void* d_out, int out_size, void* d_ws, size_t ws_size,
                              hipStream_t stream) {
    (void)n_in; (void)out_size; (void)ws_size;
    const float* x    = (const float*)d_in[0];
    const float* wl   = (const float*)d_in[36];
    const float* bl   = (const float*)d_in[37];
    const int*   kptr = (const int*)d_in[38];
    int B = in_sizes[0] / (28 * 28);

    // --- bump allocator on workspace (recomputed each call; no retained state)
    char* ws = (char*)d_ws;
    size_t off = 0;
    auto alloc = [&](size_t bytes) -> void* {
        void* p = ws + off;
        off = align_up(off + bytes);
        return p;
    };

    // activations
    float* conv1 = (float*)alloc((size_t)B * 10 * 28 * 28 * 4);
    float* pool1 = (float*)alloc((size_t)B * 10 * 14 * 14 * 4);
    float* conv2 = (float*)alloc((size_t)B * 10 * 14 * 14 * 4);
    float* pool2 = (float*)alloc((size_t)B * 10 *  7 *  7 * 4);
    float* conv3 = (float*)alloc((size_t)B * 20 *  7 *  7 * 4);
    float* cat4  = (float*)alloc((size_t)B * 30 * 14 * 14 * 4);
    float* conv4 = (float*)alloc((size_t)B * 10 * 14 * 14 * 4);
    float* cat5  = (float*)alloc((size_t)B * 20 * 28 * 28 * 4);
    float* conv5 = (float*)alloc((size_t)B * 10 * 28 * 28 * 4);

    // per-layer gate/weight scratch (Kpad max = 288, Mpad max = 32)
    MoeScratch sc[5];
    for (int l = 0; l < 5; ++l) {
        sc[l].Ap   = (_Float16*)alloc((size_t)4 * 32 * 288 * 2);
        sc[l].graw = (float*)alloc((size_t)B * 4 * 4);
        sc[l].mask = (float*)alloc(4 * 4);
        sc[l].gate = (float*)alloc((size_t)B * 4 * 4);
    }

    // ---- layer 1: [B,1,28,28] -> conv1 [B,10,28,28]
    run_moe_layer(x, B, 1, 28, 28, 10, d_in, 1, kptr, conv1, sc[0], stream);
    {   size_t tot = (size_t)B * 10 * 14 * 14;
        maxpool2_kernel<<<(int)((tot + 255) / 256), 256, 0, stream>>>(conv1, pool1, B, 10, 28, 28); }

    // ---- layer 2: [B,10,14,14] -> conv2 [B,10,14,14]
    run_moe_layer(pool1, B, 10, 14, 14, 10, d_in, 8, kptr, conv2, sc[1], stream);
    {   size_t tot = (size_t)B * 10 * 7 * 7;
        maxpool2_kernel<<<(int)((tot + 255) / 256), 256, 0, stream>>>(conv2, pool2, B, 10, 14, 14); }

    // ---- layer 3: [B,10,7,7] -> conv3 [B,20,7,7]
    run_moe_layer(pool2, B, 10, 7, 7, 20, d_in, 15, kptr, conv3, sc[2], stream);

    // ---- concat [up2(conv3), conv2] -> cat4 [B,30,14,14]
    {   size_t tot = (size_t)B * 20 * 14 * 14;
        upsample2_kernel<<<(int)((tot + 255) / 256), 256, 0, stream>>>(conv3, cat4, B, 20, 7, 7, 30, 0);
        size_t tot2 = (size_t)B * 10 * 14 * 14;
        copy_ch_kernel<<<(int)((tot2 + 255) / 256), 256, 0, stream>>>(conv2, cat4, B, 10, 14, 14, 30, 20); }

    // ---- layer 4: [B,30,14,14] -> conv4 [B,10,14,14]
    run_moe_layer(cat4, B, 30, 14, 14, 10, d_in, 22, kptr, conv4, sc[3], stream);

    // ---- concat [up2(conv4), conv1] -> cat5 [B,20,28,28]
    {   size_t tot = (size_t)B * 10 * 28 * 28;
        upsample2_kernel<<<(int)((tot + 255) / 256), 256, 0, stream>>>(conv4, cat5, B, 10, 14, 14, 20, 0);
        copy_ch_kernel<<<(int)((tot + 255) / 256), 256, 0, stream>>>(conv1, cat5, B, 10, 28, 28, 20, 10); }

    // ---- layer 5: [B,20,28,28] -> conv5 [B,10,28,28]
    run_moe_layer(cat5, B, 20, 28, 28, 10, d_in, 29, kptr, conv5, sc[4], stream);

    // ---- final 1x1 projection -> d_out [B,1,28,28]
    {   size_t tot = (size_t)B * 28 * 28;
        final_proj_kernel<<<(int)((tot + 255) / 256), 256, 0, stream>>>(
            conv5, wl, bl, (float*)d_out, B, 10, 28 * 28); }
}
